// Feat_Merger_75342316306689
// MI455X (gfx1250) — compile-verified, matrix operand order checked by reading, not run on hardware
//
#include <hip/hip_runtime.h>
#include <hip/hip_bf16.h>
#include <stdint.h>

#define D    768
#define EPS  1e-5f

typedef __attribute__((ext_vector_type(16))) _Float16 v16h;
typedef __attribute__((ext_vector_type(8)))  _Float16 v8h;
typedef __attribute__((ext_vector_type(4)))  _Float16 v4h;
typedef __attribute__((ext_vector_type(8)))  float    v8f;
typedef __attribute__((ext_vector_type(4)))  float    v4f;

__device__ __forceinline__ v8f wmma32(v16h a, v16h b, v8f c) {
  // D = A(16x32 f16) x B(32x16 f16) + C(16x16 f32)
  return __builtin_amdgcn_wmma_f32_16x16x32_f16(false, a, false, b, (short)0, c,
                                                false, false);
}

// Load one 16x32 f16 fragment from row-major memory with leading dim ld.
// Valid as A (row-major MxK) and as B when the source is W stored (N,K)
// row-major (i.e. B = W^T). Matches CDNA5 16-bit A layout:
//   lane<16 : row=lane,    K runs {0..7} and {16..23}
//   lane>=16: row=lane-16, K runs {8..15} and {24..31}
__device__ __forceinline__ v16h load_frag(const _Float16* base, int ld, int lane) {
  const int row  = lane & 15;
  const int koff = (lane >> 4) << 3;
  const _Float16* p = base + (size_t)row * ld + koff;
  v8h lo = *(const v8h*)(p);
  v8h hi = *(const v8h*)(p + 16);
  v16h r;
#pragma unroll
  for (int i = 0; i < 8; ++i) { r[i] = lo[i]; r[i + 8] = hi[i]; }
  return r;
}

__device__ __forceinline__ float gelu_exact(float x) {
  return 0.5f * x * (1.0f + erff(x * 0.70710678118654752440f));
}

// ---------------------------------------------------------------- f32 -> f16
__global__ __launch_bounds__(256) void k_f32_to_f16(const float* __restrict__ s,
                                                    _Float16* __restrict__ d,
                                                    int n) {
  const int i = (blockIdx.x * 256 + threadIdx.x) * 4;
  if (i < n) {  // n is always a multiple of 4 here
    v4f v = *(const v4f*)(s + i);
    v4h h;
#pragma unroll
    for (int k = 0; k < 4; ++k) h[k] = (_Float16)v[k];
    *(v4h*)(d + i) = h;
  }
}

// ------------------------------------- depthwise conv + GELU + mask-mean + LN
__global__ __launch_bounds__(256) void k_conv_mean_ln(
    const float* __restrict__ speech, const int* __restrict__ lens,
    const float* __restrict__ cw, const float* __restrict__ cb,
    const float* __restrict__ gamma, const float* __restrict__ beta,
    float* __restrict__ agg32, _Float16* __restrict__ agg16) {
  __shared__ float red[256], red2[256], smv[2];
  const int b = blockIdx.x, tid = threadIdx.x;
  int len = lens[b];
  len = (len < 1) ? 1 : ((len > 191) ? 191 : len);

  float w[3][20], bia[3], s[3] = {0.f, 0.f, 0.f};
#pragma unroll
  for (int ci = 0; ci < 3; ++ci) {
    const int ch = tid + ci * 256;
    bia[ci] = cb[ch];
#pragma unroll
    for (int k = 0; k < 20; ++k) w[ci][k] = cw[ch * 20 + k];
  }
  const float* sp = speech + (size_t)b * 400 * D;
  for (int t = 0; t < len; ++t) {
    if (t + 2 < len) {  // stream ahead: gfx1250 global_prefetch_b8
      const float* nrow = sp + (size_t)(2 * (t + 2)) * D;
      __builtin_prefetch(nrow + tid, 0, 1);
      __builtin_prefetch(nrow + tid + 256, 0, 1);
      __builtin_prefetch(nrow + tid + 512, 0, 1);
    }
    float a0 = bia[0], a1 = bia[1], a2 = bia[2];
#pragma unroll
    for (int k = 0; k < 20; ++k) {
      const float* row = sp + (size_t)(2 * t + k) * D;
      a0 += row[tid]       * w[0][k];
      a1 += row[tid + 256] * w[1][k];
      a2 += row[tid + 512] * w[2][k];
    }
    s[0] += gelu_exact(a0); s[1] += gelu_exact(a1); s[2] += gelu_exact(a2);
  }
  const float invl = 1.f / (float)len;
  const float v0 = s[0] * invl, v1 = s[1] * invl, v2 = s[2] * invl;
  red[tid]  = v0 + v1 + v2;
  red2[tid] = v0 * v0 + v1 * v1 + v2 * v2;
  __syncthreads();
  for (int off = 128; off > 0; off >>= 1) {
    if (tid < off) { red[tid] += red[tid + off]; red2[tid] += red2[tid + off]; }
    __syncthreads();
  }
  if (tid == 0) {
    const float m = red[0] * (1.f / 768.f);
    const float var = red2[0] * (1.f / 768.f) - m * m;
    smv[0] = m; smv[1] = rsqrtf(var + EPS);
  }
  __syncthreads();
  const float m = smv[0], rstd = smv[1];
  const size_t base = (size_t)b * D;
  float y;
  y = (v0 - m) * rstd * gamma[tid] + beta[tid];
  agg32[base + tid] = y; agg16[base + tid] = (_Float16)y;
  y = (v1 - m) * rstd * gamma[tid + 256] + beta[tid + 256];
  agg32[base + tid + 256] = y; agg16[base + tid + 256] = (_Float16)y;
  y = (v2 - m) * rstd * gamma[tid + 512] + beta[tid + 512];
  agg32[base + tid + 512] = y; agg16[base + tid + 512] = (_Float16)y;
}

// ------------------- 16-row GEMM block: Y = LN(X W^T + b [+resid]) [+resid]
// RMODE: 0 = no residual, 1 = residual added before LN, 2 = added after LN.
template <int RMODE, bool OUT32>
__global__ __launch_bounds__(256) void k_gemm_ln(
    const _Float16* __restrict__ X16, const float* __restrict__ resid,
    const _Float16* __restrict__ W16, const float* __restrict__ bias,
    const float* __restrict__ gamma, const float* __restrict__ beta,
    float* __restrict__ out32, _Float16* __restrict__ out16) {
  __shared__ float pre[16][D];
  __shared__ float ps[16][16], ps2[16][16];
  __shared__ float rm[16], rs[16];
  const int tid = threadIdx.x, wave = tid >> 5, lane = tid & 31;
  const int row0 = blockIdx.x * 16;
  const int nbase = wave * 96;            // 6 tiles * 16 cols per wave
  const int ncl = lane & 15, rb = (lane >> 4) * 8;

#pragma unroll
  for (int i = 0; i < 6; ++i)
    __builtin_prefetch(W16 + (size_t)(nbase + i * 16) * D, 0, 1);

  v8f acc[6] = {};
#pragma unroll 2
  for (int kb = 0; kb < D; kb += 32) {
    // Issue A + all six B fragment loads first (in-order completion), then
    // consume in issue order -> partial s_wait_loadcnt, latency overlapped.
    v16h a = load_frag(X16 + (size_t)row0 * D + kb, D, lane);
    v16h bfr[6];
#pragma unroll
    for (int i = 0; i < 6; ++i)
      bfr[i] = load_frag(W16 + (size_t)(nbase + i * 16) * D + kb, D, lane);
#pragma unroll
    for (int i = 0; i < 6; ++i) acc[i] = wmma32(a, bfr[i], acc[i]);
  }
#pragma unroll
  for (int i = 0; i < 6; ++i) {
    const int ncol = nbase + i * 16 + ncl;
    const float bb = bias[ncol];
#pragma unroll
    for (int r = 0; r < 8; ++r) pre[rb + r][ncol] = acc[i][r] + bb;
  }
  __syncthreads();

  const int lrow = tid >> 4, c0 = tid & 15;
  float vloc[48];
  float sa = 0.f, sb = 0.f;
#pragma unroll
  for (int j = 0; j < 48; ++j) {
    float v = pre[lrow][c0 + 16 * j];
    if (RMODE == 1) v += resid[(size_t)(row0 + lrow) * D + c0 + 16 * j];
    vloc[j] = v; sa += v; sb += v * v;
  }
  ps[lrow][c0] = sa; ps2[lrow][c0] = sb;
  __syncthreads();
  if (tid < 16) {
    float a = 0.f, b2 = 0.f;
    for (int j = 0; j < 16; ++j) { a += ps[tid][j]; b2 += ps2[tid][j]; }
    const float m = a * (1.f / 768.f);
    const float var = b2 * (1.f / 768.f) - m * m;
    rm[tid] = m; rs[tid] = rsqrtf(var + EPS);
  }
  __syncthreads();
  const float m = rm[lrow], rstd = rs[lrow];
#pragma unroll
  for (int j = 0; j < 48; ++j) {
    const int c = c0 + 16 * j;
    float y = (vloc[j] - m) * rstd * gamma[c] + beta[c];
    if (RMODE == 2) y += resid[(size_t)(row0 + lrow) * D + c];
    if (OUT32) out32[(size_t)(row0 + lrow) * D + c] = y;
    out16[(size_t)(row0 + lrow) * D + c] = (_Float16)y;
  }
}

// --------------------------- single-head attention per segment (16 queries)
// KV = 16 (self) or 32 (cross). out gets LN(resid + attn_out) (resid optional).
template <int KV, bool HAS_RESID>
__global__ __launch_bounds__(256) void k_attn(
    const _Float16* __restrict__ q16, const float* __restrict__ resid32,
    const _Float16* __restrict__ kv16,
    const _Float16* __restrict__ Wq, const float* __restrict__ bq,
    const _Float16* __restrict__ Wk, const float* __restrict__ bk,
    const _Float16* __restrict__ Wv, const float* __restrict__ bv,
    const _Float16* __restrict__ Wo, const float* __restrict__ bo,
    const float* __restrict__ gamma, const float* __restrict__ beta,
    float* __restrict__ out) {
  __shared__ _Float16 bufA[16 * D];     // Q (phase 1-2) then OUT (phase 3-4)
  __shared__ _Float16 bufB[8][1024];    // per-wave K/V staging
  __shared__ float    sc[16][32];       // score accumulators
  __shared__ _Float16 at[16][32];       // softmaxed attention (f16)
  __shared__ float    ps[16][16], ps2[16][16];
  __shared__ float    rm[16], rs[16];

  const int tid = threadIdx.x, wave = tid >> 5, lane = tid & 31;
  const int nbase = wave * 96;
  const int ncl = lane & 15, rb = (lane >> 4) * 8;
  const _Float16* qsrc = q16 + (size_t)blockIdx.x * 16 * D;
  const _Float16* ksrc = kv16 + (size_t)blockIdx.x * KV * D;

#pragma unroll
  for (int i = 0; i < 6; ++i) {
    __builtin_prefetch(Wq + (size_t)(nbase + i * 16) * D, 0, 1);
    __builtin_prefetch(Wo + (size_t)(nbase + i * 16) * D, 0, 1);
  }

  // ---- Phase 1: Q = qsrc Wq^T + bq  -> bufA (f16)
  {
    v8f qacc[6] = {};
#pragma unroll 2
    for (int kb = 0; kb < D; kb += 32) {
      v16h a = load_frag(qsrc + kb, D, lane);
      v16h bfr[6];
#pragma unroll
      for (int i = 0; i < 6; ++i)
        bfr[i] = load_frag(Wq + (size_t)(nbase + i * 16) * D + kb, D, lane);
#pragma unroll
      for (int i = 0; i < 6; ++i) qacc[i] = wmma32(a, bfr[i], qacc[i]);
    }
#pragma unroll
    for (int i = 0; i < 6; ++i) {
      const int n0 = nbase + i * 16;
      const float bb = bq[n0 + ncl];
#pragma unroll
      for (int r = 0; r < 8; ++r)
        bufA[(size_t)(rb + r) * D + n0 + ncl] = (_Float16)(qacc[i][r] + bb);
    }
  }
  for (int idx = tid; idx < 16 * 32; idx += 256) ((float*)sc)[idx] = 0.f;
  __syncthreads();

  // ---- Phase 2: scores = Q K^T, K-tiles computed on the fly
  {
    _Float16* Kst = &bufB[wave][0];  // [16 kv][32 ch]
#pragma unroll
    for (int kvg = 0; kvg < KV / 16; ++kvg) {
      v8f accS = {};
      for (int si = 0; si < 3; ++si) {
        const int kb32 = (wave * 3 + si) * 32;
        v8f ak0 = {}, ak1 = {};
#pragma unroll 2
        for (int kb = 0; kb < D; kb += 32) {
          v16h a  = load_frag(ksrc + (size_t)kvg * 16 * D + kb, D, lane);
          v16h b0 = load_frag(Wk + (size_t)kb32 * D + kb, D, lane);
          v16h b1 = load_frag(Wk + (size_t)(kb32 + 16) * D + kb, D, lane);
          ak0 = wmma32(a, b0, ak0);
          ak1 = wmma32(a, b1, ak1);
        }
        const float bb0 = bk[kb32 + ncl];
        const float bb1 = bk[kb32 + 16 + ncl];
#pragma unroll
        for (int r = 0; r < 8; ++r) {
          Kst[(rb + r) * 32 + ncl]      = (_Float16)(ak0[r] + bb0);
          Kst[(rb + r) * 32 + 16 + ncl] = (_Float16)(ak1[r] + bb1);
        }
        v16h aq  = load_frag(bufA + kb32, D, lane);  // Q[:, kb32:kb32+32]
        v16h bkf = load_frag(Kst, 32, lane);          // (K chunk)^T
        accS = wmma32(aq, bkf, accS);
      }
#pragma unroll
      for (int r = 0; r < 8; ++r)
        atomicAdd(&sc[rb + r][kvg * 16 + ncl], accS[r]);
    }
  }
  __syncthreads();

  // ---- softmax (scale 1/sqrt(768)) -> at (f16), zero-padded to 32 cols
  if (tid < 16) {
    const float scale = 0.036084391824351615f;  // 1/sqrt(768)
    float mx = -1e30f;
    for (int c = 0; c < KV; ++c) mx = fmaxf(mx, sc[tid][c] * scale);
    float e[KV];
    float ssum = 0.f;
    for (int c = 0; c < KV; ++c) { e[c] = __expf(sc[tid][c] * scale - mx); ssum += e[c]; }
    const float inv = 1.f / ssum;
    for (int c = 0; c < 32; ++c)
      at[tid][c] = (_Float16)((c < KV) ? e[c] * inv : 0.f);
  }
  __syncthreads();

  // ---- Phase 3: out = attn @ V, V-tiles computed on the fly, staged transposed
  for (int i = 0; i < 6; ++i) {
    const int n0 = nbase + i * 16;
    _Float16* Vst = &bufB[wave][0];  // [16 ch][32 kv]
    if (KV == 16) {
      for (int idx = lane; idx < 16 * 32; idx += 32)
        if ((idx & 31) >= 16) Vst[idx] = (_Float16)0.f;
    }
#pragma unroll
    for (int kvg = 0; kvg < KV / 16; ++kvg) {
      v8f acc = {};
#pragma unroll 4
      for (int kb = 0; kb < D; kb += 32) {
        v16h a = load_frag(ksrc + (size_t)kvg * 16 * D + kb, D, lane);
        v16h b = load_frag(Wv + (size_t)n0 * D + kb, D, lane);
        acc = wmma32(a, b, acc);
      }
      const float bb = bv[n0 + ncl];
#pragma unroll
      for (int r = 0; r < 8; ++r)
        Vst[ncl * 32 + kvg * 16 + rb + r] = (_Float16)(acc[r] + bb);  // V^T
    }
    v16h aa = load_frag(&at[0][0], 32, lane);
    v16h bb = load_frag(Vst, 32, lane);
    v8f accO = {};
    accO = wmma32(aa, bb, accO);
#pragma unroll
    for (int r = 0; r < 8; ++r)
      bufA[(size_t)(rb + r) * D + n0 + ncl] = (_Float16)accO[r];  // OUT (f16)
  }
  __syncthreads();

  // ---- Phase 4: Wo projection + bias, then residual + row LN on write-back
  float* orow = out + (size_t)blockIdx.x * 16 * D;
  {
    v8f oacc[6] = {};
#pragma unroll 2
    for (int kb = 0; kb < D; kb += 32) {
      v16h a = load_frag(bufA + kb, D, lane);
      v16h bfr[6];
#pragma unroll
      for (int i = 0; i < 6; ++i)
        bfr[i] = load_frag(Wo + (size_t)(nbase + i * 16) * D + kb, D, lane);
#pragma unroll
      for (int i = 0; i < 6; ++i) oacc[i] = wmma32(a, bfr[i], oacc[i]);
    }
#pragma unroll
    for (int i = 0; i < 6; ++i) {
      const int ncol = nbase + i * 16 + ncl;
      const float bb = bo[ncol];
#pragma unroll
      for (int r = 0; r < 8; ++r)
        orow[(size_t)(rb + r) * D + ncol] = oacc[i][r] + bb;
    }
  }
  __threadfence();
  __syncthreads();
  {
    const int lrow = tid >> 4, c0 = tid & 15;
    float vloc[48];
    float sa = 0.f, sb = 0.f;
#pragma unroll
    for (int j = 0; j < 48; ++j) {
      const int c = c0 + 16 * j;
      float v = orow[(size_t)lrow * D + c];
      if (HAS_RESID) v += resid32[(size_t)(blockIdx.x * 16 + lrow) * D + c];
      vloc[j] = v; sa += v; sb += v * v;
    }
    ps[lrow][c0] = sa; ps2[lrow][c0] = sb;
    __syncthreads();
    if (tid < 16) {
      float a = 0.f, b2 = 0.f;
      for (int j = 0; j < 16; ++j) { a += ps[tid][j]; b2 += ps2[tid][j]; }
      const float m = a * (1.f / 768.f);
      const float var = b2 * (1.f / 768.f) - m * m;
      rm[tid] = m; rs[tid] = rsqrtf(var + EPS);
    }
    __syncthreads();
    const float m = rm[lrow], rstd = rs[lrow];
#pragma unroll
    for (int j = 0; j < 48; ++j) {
      const int c = c0 + 16 * j;
      orow[(size_t)lrow * D + c] = (vloc[j] - m) * rstd * gamma[c] + beta[c];
    }
  }
}

// ---------------------------------------------------------------------------
extern "C" void kernel_launch(void* const* d_in, const int* in_sizes, int n_in,
                              void* d_out, int out_size, void* d_ws, size_t ws_size,
                              hipStream_t stream) {
  (void)in_sizes; (void)n_in; (void)out_size;
  // Input order: speech, text, segment_len, chunk_timestep_len, max_pad,
  // params{conv_w, conv_b, sa{Wq,bq,Wk,bk,Wv,bv,Wo,bo}, ca{...},
  //        W_text,b_text, W_sp,b_sp, W_sp2,b_sp2,
  //        g_tn,b_tn,g_sn,b_sn,g_san,b_san,g_can,b_can,g_n1,b_n1,g_n2,b_n2}
  const float* speech = (const float*)d_in[0];
  const float* text   = (const float*)d_in[1];
  const int*   ctl    = (const int*)d_in[3];
  const float* conv_w = (const float*)d_in[5];
  const float* conv_b = (const float*)d_in[6];

  uint8_t* w = (uint8_t*)d_ws;
  size_t used = 0;
  auto take = [&](size_t n) -> uint8_t* {
    uint8_t* p = w; size_t a = (n + 255) & ~(size_t)255; w += a; used += a; return p;
  };
  const size_t WB = (size_t)D * D * 2;
  _Float16* Wsp16   = (_Float16*)take(WB);
  _Float16* Wsp2_16 = (_Float16*)take(WB);
  _Float16* Wtx16   = (_Float16*)take(WB);
  _Float16* saWq16  = (_Float16*)take(WB);
  _Float16* saWk16  = (_Float16*)take(WB);
  _Float16* saWv16  = (_Float16*)take(WB);
  _Float16* saWo16  = (_Float16*)take(WB);
  _Float16* caWq16  = (_Float16*)take(WB);
  _Float16* caWk16  = (_Float16*)take(WB);
  _Float16* caWv16  = (_Float16*)take(WB);
  _Float16* caWo16  = (_Float16*)take(WB);
  _Float16* text16  = (_Float16*)take((size_t)512 * D * 2);
  float*    agg32   = (float*)take((size_t)256 * D * 4);
  _Float16* agg16   = (_Float16*)take((size_t)256 * D * 2);
  _Float16* s1n16   = (_Float16*)take((size_t)256 * D * 2);
  float*    s32     = (float*)take((size_t)256 * D * 4);
  _Float16* s16     = (_Float16*)take((size_t)256 * D * 2);
  _Float16* tf16    = (_Float16*)take((size_t)512 * D * 2);
  if (used > ws_size) return;  // scratch too small: bail deterministically

  const int WN = D * D;
  const int gW = (WN / 4 + 255) / 256;
  k_f32_to_f16<<<gW, 256, 0, stream>>>((const float*)d_in[25], Wsp16,   WN);
  k_f32_to_f16<<<gW, 256, 0, stream>>>((const float*)d_in[27], Wsp2_16, WN);
  k_f32_to_f16<<<gW, 256, 0, stream>>>((const float*)d_in[23], Wtx16,   WN);
  k_f32_to_f16<<<gW, 256, 0, stream>>>((const float*)d_in[7],  saWq16,  WN);
  k_f32_to_f16<<<gW, 256, 0, stream>>>((const float*)d_in[9],  saWk16,  WN);
  k_f32_to_f16<<<gW, 256, 0, stream>>>((const float*)d_in[11], saWv16,  WN);
  k_f32_to_f16<<<gW, 256, 0, stream>>>((const float*)d_in[13], saWo16,  WN);
  k_f32_to_f16<<<gW, 256, 0, stream>>>((const float*)d_in[15], caWq16,  WN);
  k_f32_to_f16<<<gW, 256, 0, stream>>>((const float*)d_in[17], caWk16,  WN);
  k_f32_to_f16<<<gW, 256, 0, stream>>>((const float*)d_in[19], caWv16,  WN);
  k_f32_to_f16<<<gW, 256, 0, stream>>>((const float*)d_in[21], caWo16,  WN);
  k_f32_to_f16<<<(512 * D / 4 + 255) / 256, 256, 0, stream>>>(text, text16, 512 * D);

  // conv + GELU + masked mean + LN(n1)
  k_conv_mean_ln<<<256, 256, 0, stream>>>(speech, ctl, conv_w, conv_b,
      (const float*)d_in[37], (const float*)d_in[38], agg32, agg16);

  // s1 = LN_sn(agg + agg Wsp^T + b)      (resid pre-LN)
  k_gemm_ln<1, false><<<16, 256, 0, stream>>>(agg16, agg32, Wsp16,
      (const float*)d_in[26], (const float*)d_in[31], (const float*)d_in[32],
      nullptr, s1n16);
  // s  = LN_n2(s1 Wsp2^T + b)
  k_gemm_ln<0, true><<<16, 256, 0, stream>>>(s1n16, nullptr, Wsp2_16,
      (const float*)d_in[28], (const float*)d_in[39], (const float*)d_in[40],
      s32, s16);
  // t_feat = text + LN_tn(text Wtx^T + b)  (resid post-LN)
  k_gemm_ln<2, false><<<32, 256, 0, stream>>>(text16, text, Wtx16,
      (const float*)d_in[24], (const float*)d_in[29], (const float*)d_in[30],
      nullptr, tf16);

  float* out = (float*)d_out;
  // self-attention: LN_san(s_feat + MHA(s,s,s))
  k_attn<16, true><<<16, 256, 0, stream>>>(s16, s32, s16,
      saWq16, (const float*)d_in[8],  saWk16, (const float*)d_in[10],
      saWv16, (const float*)d_in[12], saWo16, (const float*)d_in[14],
      (const float*)d_in[33], (const float*)d_in[34], out);
  // cross-attention: LN_can(MHA(s, t_feat, t_feat))
  k_attn<32, false><<<16, 256, 0, stream>>>(s16, nullptr, tf16,
      caWq16, (const float*)d_in[16], caWk16, (const float*)d_in[18],
      caWv16, (const float*)d_in[20], caWo16, (const float*)d_in[22],
      (const float*)d_in[35], (const float*)d_in[36], out + (size_t)16 * 16 * D);
}
